// RFU_75385265979633
// MI455X (gfx1250) — matrix operand order repaired and checked by reading, not compiled, not run on hardware
//
#include <hip/hip_runtime.h>

// ---------------------------------------------------------------------------
// MI455X (gfx1250) pipeline:
//   f32 -> bf16 pixel-major transpose -> banded-WMMA correlation ->
//   normalize -> assemble -> LDS-staged WMMA 1x1 conv (GEMM 8192x256x512,
//   async global->LDS copies, ds-fed fragments, A-reuse) ->
//   WMMA 3x3 conv (im2col GEMM 8192x16x2304) -> small convs -> softmax blend.
// Matrix math: v_wmma_f32_16x16x32_bf16 (f32 accumulate).
// Data movement: global_load_async_to_lds_b128 + s_wait_asynccnt (ASYNCcnt),
// global_prefetch_b8 via __builtin_prefetch.
// Register budget: B fragments are held 4-at-a-time (32 VGPRs) so the
// single-wave WMMA kernels stay under the VGPR limit (no scratch spills).
// ---------------------------------------------------------------------------

typedef __attribute__((ext_vector_type(16))) __bf16 bf16x16;
typedef __attribute__((ext_vector_type(8)))  __bf16 bf16x8;
typedef __attribute__((ext_vector_type(8)))  float  f32x8;

#define BB   2
#define CCH  256
#define HHH  64
#define WWW  64
#define HWW  4096          // H*W
#define NPIX (BB * HWW)    // 8192
#define DDISP 4
#define KW   9             // window side
#define K2   81            // displacement channels

__device__ __forceinline__ __bf16 f2bf(float x) {
  union { float f; unsigned u; } v; v.f = x;
  unsigned r = v.u + 0x7FFFu + ((v.u >> 16) & 1u);   // round-to-nearest-even
  unsigned short h = (unsigned short)(r >> 16);
  return __builtin_bit_cast(__bf16, h);
}

__device__ __forceinline__ bf16x16 bf_zero16() {
  bf16x16 z = {};
  return z;
}

// Build a 16x32 bf16 A fragment for one lane from one contiguous 16-element
// run (lane<16 holds K 0..15, lane>=16 holds K 16..31; matches B fragments).
__device__ __forceinline__ bf16x16 load_a_frag(const __bf16* row, int lhi) {
  return *(const bf16x16*)(row + (lhi ? 16 : 0));
}

__device__ __forceinline__ f32x8 wmma_bf16(bf16x16 a, bf16x16 b, f32x8 c) {
  return __builtin_amdgcn_wmma_f32_16x16x32_bf16(false, a, false, b,
                                                 (short)0, c, false, false);
}

// Async global -> LDS copy of 16 bytes (tracked by ASYNCcnt).
__device__ __forceinline__ void async_cp_b128(unsigned lds_off, const void* gp) {
  asm volatile("global_load_async_to_lds_b128 %0, %1, off"
               :: "v"(lds_off), "v"(gp) : "memory");
}
__device__ __forceinline__ void wait_asynccnt0() {
  asm volatile("s_wait_asynccnt 0x0" ::: "memory");
}

// ---------------------------------------------------------------------------
// 1) Input transpose + bf16 convert.
//    featcat: [pixel][512] bf16 (cols 0..255 = feat; 256..511 filled later
//    with the aligned features -> A matrix of the 1x1 conv GEMM).
//    refbf:   [pixel][256] bf16 (B columns of the correlation GEMM).
// ---------------------------------------------------------------------------
__global__ void k_cvt_in(const float* __restrict__ feat,
                         const float* __restrict__ ref,
                         __bf16* __restrict__ featcat,
                         __bf16* __restrict__ refbf) {
  const int p = blockIdx.x;        // pixel 0..8191
  const int c = threadIdx.x;       // channel 0..255
  const int b = p >> 12;
  const int hw = p & (HWW - 1);
  const size_t src = ((size_t)b * CCH + c) * HWW + hw;
  featcat[(size_t)p * 512 + c] = f2bf(feat[src]);
  refbf[(size_t)p * 256 + c]   = f2bf(ref[src]);
}

// Weights: w1bf [256][512] bf16 (row n contiguous in K) ; w2t [16][9][256] bf16.
__global__ void k_cvt_w(const float* __restrict__ w1,
                        const float* __restrict__ w2,
                        __bf16* __restrict__ w1bf,
                        __bf16* __restrict__ w2t) {
  const int tid = threadIdx.x;
  for (int i = tid; i < 256 * 512; i += 256) w1bf[i] = f2bf(w1[i]);
  for (int i = tid; i < 16 * 2304; i += 256) {
    const int n = i / 2304, r = i % 2304, t = r / 256, c = r % 256;
    w2t[i] = f2bf(w2[n * 2304 + c * 9 + t]);   // w2 layout [O][I][3][3]
  }
}

// ---------------------------------------------------------------------------
// 2) Correlation as banded WMMA GEMM.
//    One wave per (b, y, 16-pixel x-tile). For each dy: two 16x16 output
//    tiles over ref columns x0-4 .. x0+27, K=256 as 2 groups of 4 WMMA
//    steps (4 B fragments in flight -> no VGPR spill, loads still clause).
// ---------------------------------------------------------------------------
__global__ void k_corr(const __bf16* __restrict__ featcat,
                       const __bf16* __restrict__ refbf,
                       float* __restrict__ aff) {
  __shared__ float lds[512];
  const int l   = threadIdx.x;        // 0..31 (one wave)
  const int x0  = blockIdx.x * 16;
  const int y   = blockIdx.y;
  const int b   = blockIdx.z;
  const int lm  = l & 15;
  const int lhi = (l >> 4) & 1;

  // Preload the 8 A fragments (feat rows of this tile, K=256) - reused for
  // all 9 dy iterations.
  bf16x16 afr[8];
  {
    const __bf16* arow = featcat + (size_t)(b * HWW + y * WWW + x0 + lm) * 512;
#pragma unroll
    for (int k0 = 0; k0 < 8; ++k0) afr[k0] = load_a_frag(arow + k0 * 32, lhi);
  }

  for (int dy = 0; dy < KW; ++dy) {
    const int yr = y + dy - DDISP;
    const bool yok = (yr >= 0) && (yr < HHH);
    if (yok) {
#pragma unroll
      for (int nt = 0; nt < 2; ++nt) {
        const int pos = x0 - DDISP + nt * 16 + lm;        // ref column x-pos
        const bool pok = (pos >= 0) && (pos < WWW);
        const int posc = pok ? pos : 0;
        const __bf16* brow = refbf + (size_t)(b * HWW + yr * WWW + posc) * 256;
        f32x8 acc = {};
#pragma unroll
        for (int g = 0; g < 2; ++g) {          // 2 groups x 4 K-steps
          bf16x16 bfr[4];
          if (pok) {
#pragma unroll
            for (int j = 0; j < 4; ++j)
              bfr[j] = *(const bf16x16*)(brow + (g * 4 + j) * 32 + (lhi ? 16 : 0));
          } else {
#pragma unroll
            for (int j = 0; j < 4; ++j) bfr[j] = bf_zero16();
          }
#pragma unroll
          for (int j = 0; j < 4; ++j)
            acc = wmma_bf16(afr[g * 4 + j], bfr[j], acc);
        }
        // C/D layout: vgpr r -> row r + 8*lhi, col lm.
#pragma unroll
        for (int r = 0; r < 8; ++r)
          lds[nt * 256 + (r + lhi * 8) * 16 + lm] = acc[r];
      }
    }
    __syncthreads();
    // Band extraction: aff[dy*9+dx][pixel m] = tile[m][m+dx].
    for (int t = l; t < 16 * KW; t += 32) {
      const int m  = t / KW;
      const int dx = t % KW;
      float val = 0.0f;
      if (yok) {
        const int col = m + dx;                 // 0..23
        val = lds[(col < 16 ? 0 : 256) + m * 16 + (col & 15)];
      }
      const int k = dy * KW + dx;
      aff[(((size_t)b * K2 + k) * HHH + y) * WWW + (x0 + m)] = val;
    }
    __syncthreads();
  }
}

// 3) Sum-normalize over the 81 displacement channels (in place).
__global__ void k_norm(float* __restrict__ aff) {
  const int p = blockIdx.x * blockDim.x + threadIdx.x;
  if (p >= NPIX) return;
  const int b = p >> 12;
  const int hw = p & (HWW - 1);
  float* base = aff + (size_t)b * K2 * HWW + hw;
  float s = 0.0f;
  for (int k = 0; k < K2; ++k) s += base[(size_t)k * HWW];
  const float inv = 1.0f / (s + 1e-7f);
  for (int k = 0; k < K2; ++k) base[(size_t)k * HWW] *= inv;
}

// 4) Assemble: aligned[c,p] = sum_k affn[k,p] * ref[c, p+delta_k].
//    Also writes bf16 into cols 256..511 of featcat (A of the conv1 GEMM).
__global__ void k_assemble(const float* __restrict__ aff,
                           const float* __restrict__ ref,
                           float* __restrict__ alignedf,
                           __bf16* __restrict__ featcat) {
  __shared__ float sa[K2];
  const int p = blockIdx.x;          // pixel
  const int c = threadIdx.x;         // channel
  const int b = p >> 12;
  const int hw = p & (HWW - 1);
  const int y = hw >> 6;
  const int x = hw & 63;
  if (c < K2) sa[c] = aff[((size_t)b * K2 + c) * HWW + hw];
  __syncthreads();
  const float* refc = ref + ((size_t)b * CCH + c) * HWW;
  float acc = 0.0f;
  for (int dy = 0; dy < KW; ++dy) {
    const int y2 = y + dy - DDISP;
    if (y2 < 0 || y2 >= HHH) continue;
    __builtin_prefetch(refc + (y2 + 1) * WWW + x, 0, 0);  // global_prefetch_b8
    for (int dx = 0; dx < KW; ++dx) {
      const int x2 = x + dx - DDISP;
      if (x2 < 0 || x2 >= WWW) continue;
      acc += sa[dy * KW + dx] * refc[y2 * WWW + x2];
    }
  }
  alignedf[((size_t)b * CCH + c) * HWW + hw] = acc;
  featcat[(size_t)p * 512 + 256 + c] = f2bf(acc);
}

// ---------------------------------------------------------------------------
// 5) conv1 (1x1, 512->256) as LDS-staged WMMA GEMM.
//    256-thread block (8 waves) computes a 64x64 output macro-tile.
//    K=512 staged through LDS in 4 chunks of 128 via async global->LDS
//    copies (ASYNCcnt). Each wave computes two 16x16 subtiles, reusing its
//    A fragment across both (matrix-op operand reuse).
// ---------------------------------------------------------------------------
__global__ void __launch_bounds__(256)
k_conv1(const __bf16* __restrict__ featcat,
        const __bf16* __restrict__ w1bf,
        const float* __restrict__ b1,
        __bf16* __restrict__ h1bf) {
  __shared__ alignas(32) __bf16 sA[64 * 128];   // 16 KB
  __shared__ alignas(32) __bf16 sB[64 * 128];   // 16 KB
  const int tid = threadIdx.x;
  const int l   = tid & 31;
  const int w   = tid >> 5;          // wave 0..7
  const int lm  = l & 15, lhi = (l >> 4) & 1;
  const int n0  = blockIdx.x * 64;   // output-channel macro-tile
  const int m0  = blockIdx.y * 64;   // pixel macro-tile
  const int mt  = w & 3;             // wave's m-subtile (0..3)
  const int np  = w >> 2;            // wave's n-pair (0..1)
  const unsigned aBase = (unsigned)(size_t)(void*)sA;   // LDS byte offset
  const unsigned bBase = (unsigned)(size_t)(void*)sB;

  f32x8 acc0 = {}, acc1 = {};
  for (int kc = 0; kc < 4; ++kc) {
    // Stage A/B K-chunks: 64 rows x 128 bf16 = 16 KB each, 16B per transfer.
#pragma unroll
    for (int i = 0; i < 4; ++i) {
      const int idx = tid + i * 256;           // 0..1023
      const int row = idx >> 4;
      const int c16 = idx & 15;
      const __bf16* ga = featcat + (size_t)(m0 + row) * 512 + kc * 128 + c16 * 8;
      const __bf16* gb = w1bf    + (size_t)(n0 + row) * 512 + kc * 128 + c16 * 8;
      const unsigned lo = (unsigned)(row * 256 + c16 * 16);
      async_cp_b128(aBase + lo, ga);
      async_cp_b128(bBase + lo, gb);
    }
    wait_asynccnt0();
    __syncthreads();
    // 4 WMMA K-steps on this chunk, fragments fed from LDS (ds_load).
    const __bf16* arow = sA + (mt * 16 + lm) * 128;
    const __bf16* b0r  = sB + ((np * 2 + 0) * 16 + lm) * 128;
    const __bf16* b1r  = sB + ((np * 2 + 1) * 16 + lm) * 128;
#pragma unroll
    for (int k0 = 0; k0 < 4; ++k0) {
      bf16x16 a   = load_a_frag(arow + k0 * 32, lhi);
      bf16x16 bb0 = *(const bf16x16*)(b0r + k0 * 32 + (lhi ? 16 : 0));
      bf16x16 bb1 = *(const bf16x16*)(b1r + k0 * 32 + (lhi ? 16 : 0));
      acc0 = wmma_bf16(a, bb0, acc0);
      acc1 = wmma_bf16(a, bb1, acc1);
    }
    __syncthreads();
  }
  const int nA = n0 + (np * 2 + 0) * 16 + lm;
  const int nB = n0 + (np * 2 + 1) * 16 + lm;
  const float biasA = b1[nA];
  const float biasB = b1[nB];
#pragma unroll
  for (int r = 0; r < 8; ++r) {
    const int row = m0 + mt * 16 + r + lhi * 8;
    h1bf[(size_t)row * 256 + nA] = f2bf(acc0[r] + biasA);
    h1bf[(size_t)row * 256 + nB] = f2bf(acc1[r] + biasB);
  }
}

// ---------------------------------------------------------------------------
// 6) conv2 (3x3, 256->16) as im2col WMMA GEMM: K = 9*256 = 2304 in 72 steps,
//    N = 16. Per 3x3 tap, fragments are processed in 2 groups of 4 (keeps
//    VGPR pressure ~80, no spills). Per-lane zero-padding at image borders.
// ---------------------------------------------------------------------------
__global__ void k_conv2(const __bf16* __restrict__ h1bf,
                        const __bf16* __restrict__ w2t,
                        const float* __restrict__ b2,
                        float* __restrict__ h2) {
  const int l = threadIdx.x;
  const int lm = l & 15, lhi = (l >> 4) & 1;
  const int m0 = blockIdx.x * 16;    // 16 consecutive pixels (same image row)
  const int b  = m0 >> 12;
  const int hw = m0 & (HWW - 1);
  const int y  = hw >> 6;
  const int x0 = hw & 63;
  f32x8 acc = {};
  const __bf16* brow = w2t + (size_t)lm * 2304;   // col n = lm
  for (int t = 0; t < 9; ++t) {
    const int ky = t / 3, kx = t % 3;
    const int y2 = y + ky - 1;
    if (y2 < 0 || y2 >= HHH) continue;            // uniform over the wave
    const int x2 = x0 + lm + kx - 1;
    const bool xok = (x2 >= 0) && (x2 < WWW);
    const int x2c = xok ? x2 : 0;
    const __bf16* arow = h1bf + (size_t)(b * HWW + y2 * WWW + x2c) * 256;
#pragma unroll
    for (int g = 0; g < 2; ++g) {                 // 2 groups x 4 K-steps
      bf16x16 afr[4], bfr[4];
      if (xok) {
#pragma unroll
        for (int j = 0; j < 4; ++j)
          afr[j] = load_a_frag(arow + (g * 4 + j) * 32, lhi);
      } else {
#pragma unroll
        for (int j = 0; j < 4; ++j) afr[j] = bf_zero16();
      }
#pragma unroll
      for (int j = 0; j < 4; ++j)
        bfr[j] = *(const bf16x16*)(brow + t * 256 + (g * 4 + j) * 32 + (lhi ? 16 : 0));
#pragma unroll
      for (int j = 0; j < 4; ++j) acc = wmma_bf16(afr[j], bfr[j], acc);
    }
  }
  const float bias = b2[lm];
#pragma unroll
  for (int r = 0; r < 8; ++r) {
    const int row = r + lhi * 8;
    h2[(size_t)(m0 + row) * 16 + lm] = acc[r] + bias;
  }
}

// 7) conv3 (3x3, 16->3), f32 VALU. h2 layout [pixel][16].
__global__ void k_conv3(const float* __restrict__ h2,
                        const float* __restrict__ w3,
                        const float* __restrict__ b3,
                        float* __restrict__ h3) {
  const int p = blockIdx.x * blockDim.x + threadIdx.x;
  if (p >= NPIX) return;
  const int b = p >> 12, hw = p & (HWW - 1), y = hw >> 6, x = hw & 63;
  float acc0 = b3[0], acc1 = b3[1], acc2 = b3[2];
  for (int ky = 0; ky < 3; ++ky) {
    const int y2 = y + ky - 1; if (y2 < 0 || y2 >= HHH) continue;
    for (int kx = 0; kx < 3; ++kx) {
      const int x2 = x + kx - 1; if (x2 < 0 || x2 >= WWW) continue;
      const float* hq = h2 + (size_t)(b * HWW + y2 * WWW + x2) * 16;
      for (int ic = 0; ic < 16; ++ic) {
        const float v = hq[ic];
        acc0 += v * w3[((0 * 16 + ic) * 3 + ky) * 3 + kx];
        acc1 += v * w3[((1 * 16 + ic) * 3 + ky) * 3 + kx];
        acc2 += v * w3[((2 * 16 + ic) * 3 + ky) * 3 + kx];
      }
    }
  }
  h3[(size_t)p * 3 + 0] = acc0;
  h3[(size_t)p * 3 + 1] = acc1;
  h3[(size_t)p * 3 + 2] = acc2;
}

// 8) conv4 (3x3, 3->2) + softmax -> score0.
__global__ void k_conv4sm(const float* __restrict__ h3,
                          const float* __restrict__ w4,
                          const float* __restrict__ b4,
                          float* __restrict__ score0) {
  const int p = blockIdx.x * blockDim.x + threadIdx.x;
  if (p >= NPIX) return;
  const int b = p >> 12, hw = p & (HWW - 1), y = hw >> 6, x = hw & 63;
  float l0 = b4[0], l1 = b4[1];
  for (int ky = 0; ky < 3; ++ky) {
    const int y2 = y + ky - 1; if (y2 < 0 || y2 >= HHH) continue;
    for (int kx = 0; kx < 3; ++kx) {
      const int x2 = x + kx - 1; if (x2 < 0 || x2 >= WWW) continue;
      const float* hq = h3 + (size_t)(b * HWW + y2 * WWW + x2) * 3;
      for (int ic = 0; ic < 3; ++ic) {
        const float v = hq[ic];
        l0 += v * w4[((0 * 3 + ic) * 3 + ky) * 3 + kx];
        l1 += v * w4[((1 * 3 + ic) * 3 + ky) * 3 + kx];
      }
    }
  }
  const float mx = fmaxf(l0, l1);
  const float e0 = __expf(l0 - mx), e1 = __expf(l1 - mx);
  score0[p] = e0 / (e0 + e1);
}

// 9) Blend: out = s0*feat + (1-s0)*aligned, [B,C,H,W] f32.
__global__ void k_combine(const float* __restrict__ feat,
                          const float* __restrict__ alignedf,
                          const float* __restrict__ score0,
                          float* __restrict__ out) {
  const size_t idx = (size_t)blockIdx.x * blockDim.x + threadIdx.x;
  const int b = (int)(idx >> 20);           // C*H*W = 2^20
  const int hw = (int)(idx & (HWW - 1));
  const float s = score0[b * HWW + hw];
  out[idx] = s * feat[idx] + (1.0f - s) * alignedf[idx];
}

// ---------------------------------------------------------------------------
extern "C" void kernel_launch(void* const* d_in, const int* in_sizes, int n_in,
                              void* d_out, int out_size, void* d_ws, size_t ws_size,
                              hipStream_t stream) {
  const float* feat = (const float*)d_in[0];
  const float* ref  = (const float*)d_in[1];
  const float* w1   = (const float*)d_in[2];
  const float* b1   = (const float*)d_in[3];
  const float* w2   = (const float*)d_in[4];
  const float* b2   = (const float*)d_in[5];
  const float* w3   = (const float*)d_in[6];
  const float* b3   = (const float*)d_in[7];
  const float* w4   = (const float*)d_in[8];
  const float* b4   = (const float*)d_in[9];
  float* out = (float*)d_out;

  // Workspace sub-allocation (256 B aligned): ~28 MB total.
  char* ws = (char*)d_ws;
  size_t off = 0;
  auto alloc = [&](size_t bytes) -> char* {
    char* p = ws + off;
    off = (off + bytes + 255) & ~(size_t)255;
    return p;
  };
  __bf16* featcat  = (__bf16*)alloc((size_t)NPIX * 512 * 2);   // 8 MB
  __bf16* refbf    = (__bf16*)alloc((size_t)NPIX * 256 * 2);   // 4 MB
  __bf16* w1bf     = (__bf16*)alloc((size_t)256 * 512 * 2);
  __bf16* w2t      = (__bf16*)alloc((size_t)16 * 2304 * 2);
  float*  aff      = (float*) alloc((size_t)BB * K2 * HWW * 4); // 2.6 MB
  float*  alignedf = (float*) alloc((size_t)NPIX * 256 * 4);    // 8 MB
  __bf16* h1bf     = (__bf16*)alloc((size_t)NPIX * 256 * 2);    // 4 MB
  float*  h2       = (float*) alloc((size_t)NPIX * 16 * 4);
  float*  h3       = (float*) alloc((size_t)NPIX * 3 * 4);
  float*  score0   = (float*) alloc((size_t)NPIX * 4);

  k_cvt_in  <<<NPIX, 256, 0, stream>>>(feat, ref, featcat, refbf);
  k_cvt_w   <<<1, 256, 0, stream>>>(w1, w2, w1bf, w2t);
  k_corr    <<<dim3(WWW / 16, HHH, BB), 32, 0, stream>>>(featcat, refbf, aff);
  k_norm    <<<NPIX / 256, 256, 0, stream>>>(aff);
  k_assemble<<<NPIX, 256, 0, stream>>>(aff, ref, alignedf, featcat);
  k_conv1   <<<dim3(CCH / 64, NPIX / 64), 256, 0, stream>>>(featcat, w1bf, b1, h1bf);
  k_conv2   <<<NPIX / 16, 32, 0, stream>>>(h1bf, w2t, b2, h2);
  k_conv3   <<<NPIX / 256, 256, 0, stream>>>(h2, w3, b3, h3);
  k_conv4sm <<<NPIX / 256, 256, 0, stream>>>(h3, w4, b4, score0);
  k_combine <<<NPIX * 256 / 256, 256, 0, stream>>>(feat, alignedf, score0, out);
}
